// qtSNet_54726473285816
// MI455X (gfx1250) — 8-bit matrix operand order repaired and checked by reading, compiled, not run on hardware
//
#include <hip/hip_runtime.h>
#include <stddef.h>

#define B_SZ      2048
#define K_REAL    5400
#define K_PAD     5440
#define KC_NUM    85
#define NCOL      128
#define NSTEP     10
#define NOUT      10
#define Q_STEP    0.0047058823529411764f
#define Q_INV     212.5f

#define OFF_ABSMAX 0u
#define OFF_QFC1   256u
#define OFF_M1     (256u + 696320u)
#define OFF_S1     (OFF_M1 + 11059200u)
#define OFF_M3     (OFF_S1 + 11141120u)
#define OFF_M5     (OFF_M3 + 1048576u)

typedef __attribute__((ext_vector_type(8))) int v8i;
typedef __attribute__((ext_vector_type(4))) int v4i;
typedef __attribute__((ext_vector_type(2))) int v2i;
typedef int gv4i __attribute__((vector_size(16)));

#if defined(__has_builtin)
#if __has_builtin(__builtin_amdgcn_global_load_async_to_lds_b128)
#define HAVE_ASYNC_B128 1
#endif
#if __has_builtin(__builtin_amdgcn_s_wait_asynccnt)
#define HAVE_WAIT_ASYNC 1
#endif
#endif

__device__ __forceinline__ void async_copy_b128(const void* g, unsigned ldsOff) {
#if defined(HAVE_ASYNC_B128)
    typedef __attribute__((address_space(1))) gv4i gv4i_as1;
    typedef __attribute__((address_space(3))) gv4i gv4i_as3;
    __builtin_amdgcn_global_load_async_to_lds_b128(
        (gv4i_as1*)(unsigned long long)(size_t)g, (gv4i_as3*)ldsOff, 0, 0);
#else
    unsigned long long ga = (unsigned long long)(size_t)g;
    asm volatile("global_load_async_to_lds_b128 %0, %1, off"
                 :: "v"(ldsOff), "v"(ga) : "memory");
#endif
}

__device__ __forceinline__ void wait_async4() {
#if defined(HAVE_WAIT_ASYNC)
    __builtin_amdgcn_s_wait_asynccnt(4);
#else
    asm volatile("s_wait_asynccnt 4" ::: "memory");
#endif
}

__device__ __forceinline__ void wait_async0() {
#if defined(HAVE_WAIT_ASYNC)
    __builtin_amdgcn_s_wait_asynccnt(0);
#else
    asm volatile("s_wait_asynccnt 0" ::: "memory");
#endif
}

__global__ void absmax_kernel(const float* __restrict__ w, int n, unsigned* __restrict__ out) {
    __shared__ unsigned red[256];
    unsigned m = 0u;
    for (int i = blockIdx.x * blockDim.x + threadIdx.x; i < n; i += gridDim.x * blockDim.x)
        m = max(m, __float_as_uint(fabsf(w[i])));
    red[threadIdx.x] = m;
    __syncthreads();
    for (int s = 128; s > 0; s >>= 1) {
        if ((int)threadIdx.x < s) red[threadIdx.x] = max(red[threadIdx.x], red[threadIdx.x + s]);
        __syncthreads();
    }
    if (threadIdx.x == 0) atomicMax(out, red[0]);
}

__global__ void quant_fc1_kernel(const float* __restrict__ w, const unsigned* __restrict__ absmax,
                                 signed char* __restrict__ q) {
    int idx = blockIdx.x * blockDim.x + threadIdx.x;
    if (idx >= NCOL * K_PAD) return;
    int n = idx / K_PAD, k = idx - n * K_PAD;
    float val = 0.0f;
    if (k < K_REAL) {
        float scale = __uint_as_float(absmax[1]) * (1.0f / 127.0f);
        val = fminf(fmaxf(rintf(w[n * K_REAL + k] / scale), -127.0f), 127.0f);
    }
    q[idx] = (signed char)val;
}

__global__ void lif1_kernel(const float* __restrict__ x, const float* __restrict__ wconv,
                            const unsigned* __restrict__ absmax,
                            unsigned char* __restrict__ m1, unsigned char* __restrict__ s1) {
    int idx = blockIdx.x * blockDim.x + threadIdx.x;
    if (idx >= B_SZ * K_REAL) return;
    int b = idx / K_REAL, k = idx - b * K_REAL;
    int c = k / 900, r = k - c * 900, h = r / 30, w = r - h * 30;

    float scale = __uint_as_float(absmax[0]) * (1.0f / 127.0f);
    const float* xb = x + (size_t)b * 1024;
    float cur = 0.0f;
#pragma unroll
    for (int i = 0; i < 3; ++i)
#pragma unroll
        for (int j = 0; j < 3; ++j) {
            float wt = wconv[c * 9 + i * 3 + j];
            float wq = fminf(fmaxf(rintf(wt / scale), -127.0f), 127.0f) * scale;
            cur += xb[(h + i) * 32 + (w + j)] * wq;
        }

    float mold = (float)m1[idx] * Q_STEP;
    float rst  = (mold > 1.0f) ? 1.0f : 0.0f;
    float m    = 0.9f * mold + cur - rst;
    float qi   = fminf(fmaxf(rintf(m * Q_INV), 0.0f), 255.0f);
    m1[idx]    = (unsigned char)qi;
    s1[(size_t)b * K_PAD + k] = (qi * Q_STEP > 1.0f) ? 1 : 0;
}

__device__ __forceinline__ v8i load_a(const unsigned char* aRow, int k0) {
    v2i a0 = *(const v2i*)(aRow + k0);
    v2i a1 = *(const v2i*)(aRow + k0 + 8);
    v2i a2 = *(const v2i*)(aRow + k0 + 32);
    v2i a3 = *(const v2i*)(aRow + k0 + 40);
    v8i a = {a0.x, a0.y, a1.x, a1.y, a2.x, a2.y, a3.x, a3.y};
    return a;
}

__launch_bounds__(128)
__global__ void gemm_lif_kernel(const unsigned char* __restrict__ s1,
                                const signed char*  __restrict__ qfc1,
                                const float* __restrict__ wfc3,
                                const unsigned* __restrict__ absmax,
                                float* __restrict__ m3, float* __restrict__ m5,
                                float* __restrict__ out, int t) {
    __shared__ __align__(16) unsigned char bufB[2][8192];
    __shared__ unsigned char s3_lds[64][NCOL];
    __shared__ float wq3_lds[NOUT * NCOL];

    const int tid  = threadIdx.x;
    const int lane = tid & 31;
    const int wave = tid >> 5;
    const int rowBlock = blockIdx.x * 64;
    const int rowBase  = rowBlock + wave * 16;

    const unsigned char* qfc1u = (const unsigned char*)qfc1;
    const unsigned bBase = (unsigned)(size_t)(const void*)&bufB[0][0];

#define ISSUE_B(buf, kc)                                                            \
    do {                                                                            \
        const int _k0 = (kc) * 64;                                                  \
        _Pragma("unroll")                                                           \
        for (int _i = 0; _i < 4; ++_i) {                                            \
            int _s = tid + 128 * _i;                                                \
            int _nt = _s >> 6, _rem = _s & 63, _l = _rem >> 1, _part = _rem & 1;    \
            int _col = _nt * 16 + (_l & 15), _hf = _l >> 4;                         \
            const unsigned char* _g =                                               \
                qfc1u + (size_t)_col * K_PAD + _k0 + _hf * 16 + _part * 32;         \
            unsigned _loff = bBase +                                                \
                (unsigned)((buf) * 8192 + (_nt * 32 + _l) * 32 + _part * 16);       \
            async_copy_b128(_g, _loff);                                             \
        }                                                                           \
    } while (0)

    const float scale3 = __uint_as_float(absmax[2]) * (1.0f / 127.0f);
    for (int i = tid; i < NOUT * NCOL; i += 128) {
        float w = wfc3[i];
        wq3_lds[i] = fminf(fmaxf(rintf(w / scale3), -127.0f), 127.0f) * scale3;
    }

    const int half = lane >> 4;
    const int nn   = lane & 15;

    const unsigned char* aRow = s1 + (size_t)(rowBase + nn) * K_PAD + half * 16;

    v8i acc[8];
    const v8i vzero = {0, 0, 0, 0, 0, 0, 0, 0};
#pragma unroll
    for (int i = 0; i < 8; ++i) acc[i] = vzero;

    ISSUE_B(0, 0);
    ISSUE_B(1, 1);
    v8i aCur = load_a(aRow, 0);

    for (int kc = 0; kc < KC_NUM; ++kc) {
        if (kc + 1 < KC_NUM) wait_async4();
        else                 wait_async0();
        __syncthreads();

        v8i aNext = vzero;
        if (kc + 1 < KC_NUM) aNext = load_a(aRow, (kc + 1) * 64);

        const int cur = kc & 1;
#pragma unroll
        for (int nt = 0; nt < 8; ++nt) {
            const v4i* bp = (const v4i*)&bufB[cur][(nt * 32 + lane) * 32];
            v4i b0 = bp[0];
            v4i b1 = bp[1];
            v8i b = {b0.x, b0.y, b0.z, b0.w, b1.x, b1.y, b1.z, b1.w};
            acc[nt] = __builtin_amdgcn_wmma_i32_16x16x64_iu8(false, aCur, true, b, acc[nt],
                                                             false, false);
        }
        __syncthreads();
        if (kc + 2 < KC_NUM) ISSUE_B(cur, kc + 2);
        aCur = aNext;
    }
#undef ISSUE_B

    const float scale1 = __uint_as_float(absmax[1]) * (1.0f / 127.0f);
#pragma unroll
    for (int nt = 0; nt < 8; ++nt) {
        const int col = nt * 16 + nn;
#pragma unroll
        for (int v = 0; v < 8; ++v) {
            const int mrow = v + half * 8;
            const int row  = rowBase + mrow;
            float cur  = scale1 * (float)acc[nt][v];
            float mold = m3[(size_t)row * NCOL + col];
            float rst  = (mold > 1.0f) ? 1.0f : 0.0f;
            float m    = 0.9f * mold + cur - rst;
            float qi   = fminf(fmaxf(rintf(m * Q_INV), 0.0f), 255.0f);
            float mn   = qi * Q_STEP;
            m3[(size_t)row * NCOL + col] = mn;
            s3_lds[wave * 16 + mrow][col] = (mn > 1.0f) ? 1 : 0;
        }
    }
    __syncthreads();

    for (int task = tid; task < 64 * NOUT; task += 128) {
        const int rl  = task / NOUT;
        const int j   = task - rl * NOUT;
        const int row = rowBlock + rl;
        float sum = 0.0f;
#pragma unroll 8
        for (int c = 0; c < NCOL; ++c)
            if (s3_lds[rl][c]) sum += wq3_lds[j * NCOL + c];
        float mold = m5[row * NOUT + j];
        float rst  = (mold > 1.0f) ? 1.0f : 0.0f;
        float mn   = 0.9f * mold + sum - rst;
        m5[row * NOUT + j] = mn;
        const size_t o = (size_t)t * (B_SZ * NOUT) + (size_t)row * NOUT + j;
        out[o] = (mn > 1.0f) ? 1.0f : 0.0f;
        out[(size_t)NSTEP * B_SZ * NOUT + o] = mn;
    }
}

extern "C" void kernel_launch(void* const* d_in, const int* in_sizes, int n_in,
                              void* d_out, int out_size, void* d_ws, size_t ws_size,
                              hipStream_t stream) {
    (void)in_sizes; (void)n_in; (void)out_size; (void)ws_size;
    const float* x     = (const float*)d_in[0];
    const float* wconv = (const float*)d_in[1];
    const float* wfc1  = (const float*)d_in[2];
    const float* wfc3  = (const float*)d_in[3];
    float* out = (float*)d_out;

    char* ws = (char*)d_ws;
    unsigned*      absmax = (unsigned*)(ws + OFF_ABSMAX);
    signed char*   qfc1   = (signed char*)(ws + OFF_QFC1);
    unsigned char* m1     = (unsigned char*)(ws + OFF_M1);
    unsigned char* s1     = (unsigned char*)(ws + OFF_S1);
    float*         m3     = (float*)(ws + OFF_M3);
    float*         m5     = (float*)(ws + OFF_M5);

    (void)hipMemsetAsync(absmax, 0, 3 * sizeof(unsigned), stream);
    (void)hipMemsetAsync(m1, 0, (size_t)B_SZ * K_REAL, stream);
    (void)hipMemsetAsync(s1, 0, (size_t)B_SZ * K_PAD, stream);
    (void)hipMemsetAsync(m3, 0, (size_t)B_SZ * NCOL * sizeof(float), stream);
    (void)hipMemsetAsync(m5, 0, (size_t)B_SZ * NOUT * sizeof(float), stream);

    absmax_kernel<<<1,   256, 0, stream>>>(wconv, 54,     absmax + 0);
    absmax_kernel<<<256, 256, 0, stream>>>(wfc1,  691200, absmax + 1);
    absmax_kernel<<<1,   256, 0, stream>>>(wfc3,  1280,   absmax + 2);

    quant_fc1_kernel<<<(NCOL * K_PAD) / 256, 256, 0, stream>>>(wfc1, absmax, qfc1);

    for (int t = 0; t < NSTEP; ++t) {
        lif1_kernel<<<(B_SZ * K_REAL) / 256, 256, 0, stream>>>(x, wconv, absmax, m1, s1);
        gemm_lif_kernel<<<B_SZ / 64, 128, 0, stream>>>(s1, qfc1, wfc3, absmax, m3, m5, out, t);
    }
}
